// LinearCondensed_44581760532973
// MI455X (gfx1250) — matrix operand order repaired and checked by reading, not compiled, not run on hardware
//
#include <hip/hip_runtime.h>

typedef _Float16     v16h __attribute__((ext_vector_type(16)));
typedef _Float16     h4   __attribute__((ext_vector_type(4)));
typedef float        v8f  __attribute__((ext_vector_type(8)));
typedef unsigned int v8u  __attribute__((ext_vector_type(8)));

#define B_SZ    2048
#define IN_F    4096
#define OUT_F   4096
#define FAN     32
#define BT      16                 // batch rows staged per workgroup
#define PAD_H   8                  // halves of padding -> row stride mod 64 banks = 4
#define ROW_H   (IN_F + PAD_H)     // 4104 halves per LDS row
#define WAVES   8
#define OTILES  4                  // 16-output tiles per wave
#define OUT_PER_WG (WAVES * 16 * OTILES)   // 512
#define THREADS 256

union AB16 { v8u u; v16h h; };

__global__ __launch_bounds__(THREADS)
void lincond_wmma_kernel(const float* __restrict__ x,
                         const float* __restrict__ w,
                         const float* __restrict__ bias,
                         const int*   __restrict__ idx,
                         float*       __restrict__ out)
{
    extern __shared__ _Float16 xh[];           // BT * ROW_H halves (~128 KB)

    const int tid      = threadIdx.x;
    const int bbase    = blockIdx.y * BT;
    const int obase_wg = blockIdx.x * OUT_PER_WG;

    // ---- Stage x tile (16 rows x 4096 f32) into LDS as padded f16 rows ----
    const float4* xg = reinterpret_cast<const float4*>(x + (size_t)bbase * IN_F);
    #pragma unroll 1
    for (int cid = tid; cid < BT * (IN_F / 4); cid += THREADS) {
        int r  = cid >> 10;                    // IN_F/4 = 1024 float4 chunks per row
        int c  = cid & 1023;
        float4 f = xg[(size_t)r * (IN_F / 4) + c];
        h4 h;
        h.x = (_Float16)f.x; h.y = (_Float16)f.y;
        h.z = (_Float16)f.z; h.w = (_Float16)f.w;
        *reinterpret_cast<h4*>(&xh[r * ROW_H + (c << 2)]) = h;   // 8B-aligned
    }
    __syncthreads();

    const int lane = tid & 31;
    const int wave = tid >> 5;
    const int m    = lane & 15;       // batch row within tile (A-matrix M)
    const int hi   = lane >> 4;       // selects K half per the f16 A/B VGPR layouts
    const int col  = lane & 15;       // output column for B / C / D
    // Gather base: u16 view of this lane's padded x row in LDS.
    const unsigned short* xr =
        reinterpret_cast<const unsigned short*>(xh + m * ROW_H);

    #pragma unroll 1
    for (int t = 0; t < OTILES; ++t) {
        const int obase = obase_wg + (wave * OTILES + t) * 16;

        // Prefetch next tile's indices + weights while we compute this one.
        if (t + 1 < OTILES) {
            const int onext = obase + 16;
            __builtin_prefetch(idx + (size_t)onext * FAN, 0, 1);
            __builtin_prefetch(w   + (size_t)onext * FAN, 0, 1);
        }

        // Preload this lane's one-hot B column: weights of output (obase+col),
        // K = hi*8 .. hi*8+7 and 16+hi*8 .. 16+hi*8+7, matching the A gather.
        v16h wreg;
        {
            const float* wp = w + (size_t)(obase + col) * FAN + hi * 8;
            #pragma unroll
            for (int v = 0; v < 16; ++v) wreg[v] = (_Float16)wp[v < 8 ? v : v + 8];
        }

        // Accumulator initialized with bias, broadcast along M (all 8 C VGPRs).
        float bv = bias[obase + col];
        v8f c;
        #pragma unroll
        for (int r = 0; r < 8; ++r) c[r] = bv;

        // 16 chained WMMAs in 4 groups of 4. All 16 idx int4 loads of a group
        // are hoisted first (distinct live dests -> one vmem clause, partial
        // loadcnt waits) before the 4 gather+WMMA steps consume them.
        #pragma unroll
        for (int g = 0; g < 4; ++g) {
            int4 q[4][4];
            #pragma unroll
            for (int j = 0; j < 4; ++j) {
                // Lane's K half folded into the pointer (hi*8). Needed idx are
                // the contiguous runs ipl[0..7] and ipl[16..23] (16B aligned).
                const int* ipl =
                    idx + (size_t)(obase + g * 4 + j) * FAN + hi * 8;
                q[j][0] = *reinterpret_cast<const int4*>(ipl);
                q[j][1] = *reinterpret_cast<const int4*>(ipl + 4);
                q[j][2] = *reinterpret_cast<const int4*>(ipl + 16);
                q[j][3] = *reinterpret_cast<const int4*>(ipl + 20);
            }

            #pragma unroll
            for (int j = 0; j < 4; ++j) {
                const int n = g * 4 + j;

                // 16 independent zero-extended u16 gathers (unique dests so
                // ds_loads batch and drain with partial dscnt waits).
                unsigned int g0  = xr[q[j][0].x], g1  = xr[q[j][0].y];
                unsigned int g2  = xr[q[j][0].z], g3  = xr[q[j][0].w];
                unsigned int g4  = xr[q[j][1].x], g5  = xr[q[j][1].y];
                unsigned int g6  = xr[q[j][1].z], g7  = xr[q[j][1].w];
                unsigned int g8  = xr[q[j][2].x], g9  = xr[q[j][2].y];
                unsigned int g10 = xr[q[j][2].z], g11 = xr[q[j][2].w];
                unsigned int g12 = xr[q[j][3].x], g13 = xr[q[j][3].y];
                unsigned int g14 = xr[q[j][3].z], g15 = xr[q[j][3].w];

                // Pack K pairs into dwords (v_lshl_or_b32), no subreg RMW.
                AB16 a;
                a.u[0] = g0  | (g1  << 16);
                a.u[1] = g2  | (g3  << 16);
                a.u[2] = g4  | (g5  << 16);
                a.u[3] = g6  | (g7  << 16);
                a.u[4] = g8  | (g9  << 16);
                a.u[5] = g10 | (g11 << 16);
                a.u[6] = g12 | (g13 << 16);
                a.u[7] = g14 | (g15 << 16);

                v16h z = {};
                v16h bm = (col == n) ? wreg : z;               // one-hot column

                c = __builtin_amdgcn_wmma_f32_16x16x32_f16(
                        /*neg_a=*/false, a.h, /*neg_b=*/false, bm,
                        /*c_mod=*/(short)0, c,
                        /*reuse_a=*/false, /*reuse_b=*/false);
            }
        }

        // Store D tile: element D[M][N] -> lane (N + 16*(M>=8)), VGPR r = M&7.
        float* op = out + (size_t)(bbase + 8 * hi) * OUT_F + obase + col;
        #pragma unroll
        for (int r = 0; r < 8; ++r) op[(size_t)r * OUT_F] = c[r];
    }
}

extern "C" void kernel_launch(void* const* d_in, const int* in_sizes, int n_in,
                              void* d_out, int out_size, void* d_ws, size_t ws_size,
                              hipStream_t stream)
{
    const float* x    = (const float*)d_in[0];
    const float* wgt  = (const float*)d_in[1];
    const float* bias = (const float*)d_in[2];
    const int*   idx  = (const int*)d_in[3];
    float*       out  = (float*)d_out;

    dim3 grid(OUT_F / OUT_PER_WG, B_SZ / BT);      // (8, 128) = 1024 WGs
    size_t lds_bytes = (size_t)BT * ROW_H * sizeof(_Float16);   // 131,328 B

    lincond_wmma_kernel<<<grid, dim3(THREADS), lds_bytes, stream>>>(
        x, wgt, bias, idx, out);
}
